// PositionalBias_29497835389495
// MI455X (gfx1250) — matrix operand order repaired and checked by reading, not compiled, not run on hardware
//
#include <hip/hip_runtime.h>

typedef __attribute__((ext_vector_type(16))) __bf16 v16bf;
typedef __attribute__((ext_vector_type(8)))  float  v8f;

#define L_DIM   2048
#define C_DIM   1024          // H*D = 16*64
#define N_DIM   8
#define BM      128
#define BN      128
#define BK      32
#define KSTEPS  (L_DIM / BK)  // 64
#define ASTRIDE 40            // bf16 elems per LDS row = 80 bytes (16B aligned, 20-bank pitch)

// fp32 -> bf16 round-to-nearest-even (manual, storage-only use of __bf16)
__device__ __forceinline__ unsigned short f2bf(float f) {
  unsigned u = __float_as_uint(f);
  unsigned r = u + 0x7fffu + ((u >> 16) & 1u);
  return (unsigned short)(r >> 16);
}
__device__ __forceinline__ float bf2f(unsigned short b) {
  return __uint_as_float(((unsigned)b) << 16);
}

union Frag { uint4 u[2]; v16bf v; };

// Issue one K-tile (BK x BN fp32 = 16KB) of V as async global->LDS DMA.
// Each thread moves 4 x 16B chunks; chunk ids match the convert-phase reads,
// so per-thread ASYNCcnt ordering is self-consistent.
__device__ __forceinline__ void issue_async_vtile(int t, unsigned stageBase,
                                                  const float* gRow) {
#pragma unroll
  for (int i = 0; i < 4; ++i) {
    int e     = t + i * 256;        // 16-byte chunk id, 0..1023
    int row   = e >> 5;             // k within tile, 0..31
    int col16 = e & 31;             // 16B chunk within row
    unsigned ldsOff = stageBase + (unsigned)e * 16u;
    unsigned long long ga =
        (unsigned long long)(size_t)((const char*)(gRow + (size_t)row * C_DIM) + col16 * 16);
    asm volatile("global_load_async_to_lds_b128 %0, %1, off"
                 :: "v"(ldsOff), "v"(ga) : "memory");
  }
}

__global__ __launch_bounds__(256)
void pbv_gemm_kernel(const float* __restrict__ v,
                     const float* __restrict__ w,
                     float* __restrict__ out) {
  __shared__ float4         stage[BK * BN / 4];    // 16 KB raw fp32 V tile (async dest)
  __shared__ unsigned short Ahi[BM * ASTRIDE];     // 10 KB each
  __shared__ unsigned short Alo[BM * ASTRIDE];
  __shared__ unsigned short Bhi[BN * ASTRIDE];
  __shared__ unsigned short Blo[BN * ASTRIDE];

  const int t     = threadIdx.x;
  const int lane  = t & 31;
  const int wave  = t >> 5;
  const int waveM = (wave & 3) * 32;   // 4 waves tile M
  const int waveN = (wave >> 2) * 64;  // 2 waves tile N
  const int half  = lane >> 4;
  const int l15   = lane & 15;

  const int cBase = blockIdx.x * BN;
  const int jBase = blockIdx.y * BM;
  const int n     = blockIdx.z;

  const float* vBase = v + (size_t)n * L_DIM * C_DIM + cBase;
  const unsigned stageBase = (unsigned)(size_t)(&stage[0]);  // low 32 bits = LDS byte offset

  // warm L2/L0 with w (8 KB): 256 threads x 8 floats
  __builtin_prefetch(w + t * 8, 0, 0);

  v8f acc[2][4];
#pragma unroll
  for (int mi = 0; mi < 2; ++mi)
#pragma unroll
    for (int ni = 0; ni < 4; ++ni)
#pragma unroll
      for (int r = 0; r < 8; ++r) acc[mi][ni][r] = 0.0f;

  issue_async_vtile(t, stageBase, vBase);  // prologue: K-step 0

  for (int kt = 0; kt < KSTEPS; ++kt) {
    asm volatile("s_wait_asynccnt 0" ::: "memory");
    __syncthreads();  // stage landed; prior-iter fragment reads of A/B tiles done

    // ---- generate Toeplitz A tile (bias) as bf16 hi/lo: rows j, cols k ----
    {
      const int r   = t >> 1;               // 0..127
      const int kk0 = (t & 1) * 16;
      const int j   = jBase + r;
      const int kG  = kt * BK + kk0;
#pragma unroll
      for (int kk = 0; kk < 16; ++kk) {
        int d = j - (kG + kk); d = d < 0 ? -d : d;
        float val = w[d];
        unsigned short h = f2bf(val);
        unsigned short l = f2bf(val - bf2f(h));
        Ahi[r * ASTRIDE + kk0 + kk] = h;
        Alo[r * ASTRIDE + kk0 + kk] = l;
      }
    }
    // ---- convert + transpose staged V tile into Bhi/Blo, layout [c][k] ----
    {
#pragma unroll
      for (int i = 0; i < 4; ++i) {
        int q  = t + i * 256;     // float4 id, 0..1023
        int k  = q >> 5;          // 0..31
        int c0 = (q & 31) * 4;    // 0..124
        float4 f = stage[q];
        unsigned short h0 = f2bf(f.x), h1 = f2bf(f.y), h2 = f2bf(f.z), h3 = f2bf(f.w);
        Bhi[(c0 + 0) * ASTRIDE + k] = h0;
        Bhi[(c0 + 1) * ASTRIDE + k] = h1;
        Bhi[(c0 + 2) * ASTRIDE + k] = h2;
        Bhi[(c0 + 3) * ASTRIDE + k] = h3;
        Blo[(c0 + 0) * ASTRIDE + k] = f2bf(f.x - bf2f(h0));
        Blo[(c0 + 1) * ASTRIDE + k] = f2bf(f.y - bf2f(h1));
        Blo[(c0 + 2) * ASTRIDE + k] = f2bf(f.z - bf2f(h2));
        Blo[(c0 + 3) * ASTRIDE + k] = f2bf(f.w - bf2f(h3));
      }
    }
    __syncthreads();  // tiles ready; all stage reads retired -> safe to refill stage

    if (kt + 1 < KSTEPS)
      issue_async_vtile(t, stageBase, vBase + (size_t)(kt + 1) * BK * C_DIM);  // overlaps WMMA

    // ---- load fragments (ISA 16-bit A/B VGPR layouts) + 24 WMMAs ----
    Frag ahi[2], alo[2], bhi[4], blo[4];
#pragma unroll
    for (int mi = 0; mi < 2; ++mi) {
      // A 16x32: lane row = l15; half0 K{0..7,16..23}, half1 K{8..15,24..31}
      const unsigned short* pa = &Ahi[(waveM + mi * 16 + l15) * ASTRIDE + half * 8];
      ahi[mi].u[0] = *(const uint4*)(pa);
      ahi[mi].u[1] = *(const uint4*)(pa + 16);
      const unsigned short* pl = &Alo[(waveM + mi * 16 + l15) * ASTRIDE + half * 8];
      alo[mi].u[0] = *(const uint4*)(pl);
      alo[mi].u[1] = *(const uint4*)(pl + 16);
    }
#pragma unroll
    for (int ni = 0; ni < 4; ++ni) {
      // B 32x16: lane col = l15; half0 K{0..7,16..23}, half1 K{8..15,24..31}
      const unsigned short* pb = &Bhi[(waveN + ni * 16 + l15) * ASTRIDE + half * 8];
      bhi[ni].u[0] = *(const uint4*)(pb);
      bhi[ni].u[1] = *(const uint4*)(pb + 16);
      const unsigned short* pq = &Blo[(waveN + ni * 16 + l15) * ASTRIDE + half * 8];
      blo[ni].u[0] = *(const uint4*)(pq);
      blo[ni].u[1] = *(const uint4*)(pq + 16);
    }
#pragma unroll
    for (int mi = 0; mi < 2; ++mi)
#pragma unroll
      for (int ni = 0; ni < 4; ++ni) {
        acc[mi][ni] = __builtin_amdgcn_wmma_f32_16x16x32_bf16(
            false, ahi[mi].v, false, blo[ni].v, (short)0, acc[mi][ni], false, false);
        acc[mi][ni] = __builtin_amdgcn_wmma_f32_16x16x32_bf16(
            false, alo[mi].v, false, bhi[ni].v, (short)0, acc[mi][ni], false, false);
        acc[mi][ni] = __builtin_amdgcn_wmma_f32_16x16x32_bf16(
            false, ahi[mi].v, false, bhi[ni].v, (short)0, acc[mi][ni], false, false);
      }
  }

  // ---- epilogue: C/D layout — VGPR r: lanes0-15 M=r, lanes16-31 M=r+8; N=l15 ----
  float* outBase = out + (size_t)n * L_DIM * C_DIM;
#pragma unroll
  for (int mi = 0; mi < 2; ++mi)
#pragma unroll
    for (int ni = 0; ni < 4; ++ni) {
      const int col  = cBase + waveN + ni * 16 + l15;
      const int row0 = jBase + waveM + mi * 16 + half * 8;
#pragma unroll
      for (int r = 0; r < 8; ++r)
        outBase[(size_t)(row0 + r) * C_DIM + col] = acc[mi][ni][r];
    }
}

// z_pb[i] = sum_j w[|i-j|]
__global__ __launch_bounds__(256)
void zpb_kernel(const float* __restrict__ w, float* __restrict__ z) {
  __shared__ float wl[L_DIM];
  for (int i = threadIdx.x; i < L_DIM; i += 256) wl[i] = w[i];
  __syncthreads();
  const int i = blockIdx.x * 256 + threadIdx.x;
  float s = 0.0f;
  for (int j = 0; j < L_DIM; ++j) {
    int d = i - j; d = d < 0 ? -d : d;
    s += wl[d];
  }
  z[i] = s;
}

extern "C" void kernel_launch(void* const* d_in, const int* in_sizes, int n_in,
                              void* d_out, int out_size, void* d_ws, size_t ws_size,
                              hipStream_t stream) {
  (void)in_sizes; (void)n_in; (void)out_size; (void)d_ws; (void)ws_size;
  const float* v = (const float*)d_in[0];
  const float* w = (const float*)d_in[1];
  float* out = (float*)d_out;

  dim3 grid(C_DIM / BN, L_DIM / BM, N_DIM);  // 8 x 16 x 8 = 1024 blocks
  pbv_gemm_kernel<<<grid, 256, 0, stream>>>(v, w, out);

  zpb_kernel<<<L_DIM / 256, 256, 0, stream>>>(w, out + (size_t)N_DIM * L_DIM * C_DIM);
}
